// Attention_84688165142614
// MI455X (gfx1250) — compile-verified, matrix operand order checked by reading, not run on hardware
//
#include <hip/hip_runtime.h>
#include <hip/hip_bf16.h>

// ---------------------------------------------------------------------------
// Bahdanau attention for MI455X (gfx1250, wave32, WMMA bf16)
//   B=32, S=2048, EH=512, DH=512
//   energy = tanh(enc @ w_enc + (h @ w_dec) + b);  attn = softmax(mask(energy @ v_w))
// ---------------------------------------------------------------------------

#define Bn   32
#define Sn   2048
#define Kdim 1024          // 2*EH
#define Ndim 512           // DH
#define Mdim (Bn * Sn)     // 65536

#define M_TILE   128       // rows per workgroup (16 per wave, 8 waves, 256 thr)
#define A_STRIDE 1032      // bf16 elements per LDS row (1024 + 8 pad -> bank rotate)
#define B_STRIDE 1032

typedef __attribute__((ext_vector_type(4)))  unsigned short v4us;
typedef __attribute__((ext_vector_type(8)))  unsigned short v8us;
typedef __attribute__((ext_vector_type(16))) unsigned short v16us;
typedef __attribute__((ext_vector_type(16))) __bf16        v16bf;
typedef __attribute__((ext_vector_type(8)))  float         v8f;

#define NEG_INF (-10000000000.0f)

// fp32 -> bf16 bits, round-to-nearest-even
__device__ __forceinline__ unsigned short f2bf(float f) {
    union { float f; unsigned u; } v; v.f = f;
    unsigned r = v.u + 0x7FFFu + ((v.u >> 16) & 1u);
    return (unsigned short)(r >> 16);
}

// Branch-free tanh: 1 - 2/(exp(2x)+1)  (v_exp_f32 + v_rcp_f32, no exec churn)
__device__ __forceinline__ float fast_tanh(float x) {
    float z = __builtin_amdgcn_exp2f(x * 2.885390081777927f);  // exp(2x)
    float r = __builtin_amdgcn_rcpf(z + 1.0f);
    return __builtin_fmaf(-2.0f, r, 1.0f);
}

// Load one 16x16x32 bf16 WMMA operand fragment (A or B^T row) from LDS.
// ISA layout: lanes 0-15 take K [k..k+7] and [k+16..k+23]; hi half shifted by 8.
__device__ __forceinline__ v16bf load_frag(const unsigned short* rowp, int k) {
    v8us lo = *(const v8us*)(rowp + k);
    v8us hi = *(const v8us*)(rowp + k + 16);
    v16us u = __builtin_shufflevector(lo, hi, 0,1,2,3,4,5,6,7,8,9,10,11,12,13,14,15);
    return __builtin_bit_cast(v16bf, u);
}

// ---------------------------------------------------------------------------
// Prep 1: transpose + convert w_enc (K x N fp32) -> w_encT (N x K bf16 bits)
// ---------------------------------------------------------------------------
__global__ void prep_wenc_kernel(const float* __restrict__ w_enc,
                                 unsigned short* __restrict__ wT) {
    int idx = blockIdx.x * blockDim.x + threadIdx.x;   // 512*1024 elements
    int n = idx >> 10;
    int k = idx & (Kdim - 1);
    wT[idx] = f2bf(w_enc[k * Ndim + n]);
}

// ---------------------------------------------------------------------------
// Prep 2: dec = h @ w_dec + attn_b   (32 x 512, tiny)
// ---------------------------------------------------------------------------
__global__ void prep_dec_kernel(const float* __restrict__ h,
                                const float* __restrict__ attn_w,   // rows 0..511 = w_dec
                                const float* __restrict__ attn_b,
                                float* __restrict__ dec) {
    int idx = blockIdx.x * blockDim.x + threadIdx.x;   // 32*512
    int d = idx & (Ndim - 1);
    int b = idx >> 9;
    float s = attn_b[d];
    #pragma unroll 8
    for (int k = 0; k < Ndim; ++k)
        s = fmaf(h[b * Ndim + k], attn_w[k * Ndim + d], s);
    dec[idx] = s;
}

// ---------------------------------------------------------------------------
// Main: per-WG 128-row tile of M; bf16 WMMA GEMM vs w_encT, fused tanh + v_w dot.
// Writes pre-softmax logits[m], m = s*32 + b.
// ---------------------------------------------------------------------------
__global__ void __launch_bounds__(256)
bahdanau_energy_kernel(const float* __restrict__ enc,          // [S*B, 1024] fp32
                       const unsigned short* __restrict__ wT,  // [512, 1024] bf16 bits
                       const float* __restrict__ dec,          // [32, 512]
                       const float* __restrict__ v_w,          // [512]
                       float* __restrict__ logits) {           // [65536]
    extern __shared__ char smem[];
    unsigned short* Als = (unsigned short*)smem;                           // 128 x A_STRIDE
    unsigned short* Bls = (unsigned short*)(smem + M_TILE * A_STRIDE * 2); // 16 x B_STRIDE

    const int tid  = threadIdx.x;      // 0..255
    const int wave = tid >> 5;         // 0..7
    const int lane = tid & 31;
    const int hi   = lane >> 4;        // 0 | 1
    const int l15  = lane & 15;
    const int m0   = blockIdx.x * M_TILE;
    const int myrow0 = wave * 16;

    // ---- Stage A tile: 128 x 1024 fp32 -> bf16 LDS (128 float4 chunks/thread) ----
    {
        const float4* src = (const float4*)(enc + (size_t)m0 * Kdim);
        #pragma unroll 4
        for (int i = 0; i < 128; ++i) {
            int cc  = tid + i * 256;          // 0..32767 chunk id
            int row = cc >> 8;                // 256 float4 per row
            int c4  = cc & 255;
            float4 v = src[row * 256 + c4];
            v4us p;
            p.x = f2bf(v.x); p.y = f2bf(v.y); p.z = f2bf(v.z); p.w = f2bf(v.w);
            *(v4us*)(Als + row * A_STRIDE + (c4 << 2)) = p;
        }
    }
    __syncthreads();

    float acc[8];
    #pragma unroll
    for (int j = 0; j < 8; ++j) acc[j] = 0.0f;

    // dec row pointers are block-invariant: m0 % 32 == 0  ->  b = local_row & 31
    const float* decrow[8];
    #pragma unroll
    for (int j = 0; j < 8; ++j)
        decrow[j] = dec + ((myrow0 + j + (hi << 3)) & 31) * Ndim;

    for (int nt = 0; nt < 32; ++nt) {
        const int n0 = nt * 16;

        // ---- Stage B tile: 16 x 1024 bf16 (L2 resident) ----
        {
            const v8us* srcB = (const v8us*)(wT + (size_t)n0 * Kdim);
            #pragma unroll
            for (int i = 0; i < 8; ++i) {
                int cc = tid + i * 256;        // 0..2047 chunks of 8
                int n  = cc >> 7;              // 128 chunks per row
                int c8 = cc & 127;
                v8us v = srcB[n * 128 + c8];
                *(v8us*)(Bls + n * B_STRIDE + (c8 << 3)) = v;
            }
        }
        if (nt + 1 < 32)  // warm next B tile (global_prefetch_b8)
            __builtin_prefetch(wT + (size_t)(n0 + 16) * Kdim + (tid << 2), 0, 1);
        __syncthreads();

        // ---- 16x16 output tile via 32 x v_wmma_f32_16x16x32_bf16 ----
        const unsigned short* arow = Als + (myrow0 + l15) * A_STRIDE;
        const unsigned short* brow = Bls + l15 * B_STRIDE;
        v8f c = {};
        #pragma unroll 4
        for (int kk = 0; kk < Kdim; kk += 32) {
            int ka = kk + (hi << 3);
            v16bf af = load_frag(arow, ka);
            v16bf bf = load_frag(brow, ka);
            c = __builtin_amdgcn_wmma_f32_16x16x32_bf16(
                    false, af, false, bf, (short)0, c, false, false);
        }

        // ---- fused epilogue: tanh(gemm + dec) * v_w, accumulate per row ----
        const int nn = n0 + l15;
        const float vw = v_w[nn];
        #pragma unroll
        for (int j = 0; j < 8; ++j) {
            float e = fast_tanh(c[j] + decrow[j][nn]);   // C layout: VGPR j -> M = j + 8*hi
            acc[j] = __builtin_fmaf(e, vw, acc[j]);
        }
        __syncthreads();
    }

    // ---- reduce over N lanes (xor 1,2,4,8 stays within each 16-lane half) ----
    #pragma unroll
    for (int j = 0; j < 8; ++j) {
        float v = acc[j];
        v += __shfl_xor(v, 1, 32);
        v += __shfl_xor(v, 2, 32);
        v += __shfl_xor(v, 4, 32);
        v += __shfl_xor(v, 8, 32);
        acc[j] = v;
    }
    if (l15 == 0) {
        #pragma unroll
        for (int j = 0; j < 8; ++j) {
            int m = m0 + myrow0 + j + (hi << 3);
            logits[m] = acc[j];
        }
    }
}

// ---------------------------------------------------------------------------
// Softmax over S per batch row, with mask (one block per b)
// ---------------------------------------------------------------------------
__global__ void __launch_bounds__(256)
softmax_kernel(const float* __restrict__ logits,   // [65536], index s*32+b
               const int* __restrict__ mask,       // [32, 2048]
               float* __restrict__ out) {          // [32, 2048]
    const int b   = blockIdx.x;
    const int tid = threadIdx.x;                   // 256 threads, 8 waves
    __shared__ float red[8];

    float vals[8];
    float mx = -3.4e38f;
    #pragma unroll
    for (int i = 0; i < 8; ++i) {
        int s = tid + i * 256;
        float l = (mask[b * Sn + s] == 0) ? NEG_INF : logits[s * Bn + b];
        vals[i] = l;
        mx = fmaxf(mx, l);
    }
    #pragma unroll
    for (int o = 16; o > 0; o >>= 1) mx = fmaxf(mx, __shfl_xor(mx, o, 32));
    if ((tid & 31) == 0) red[tid >> 5] = mx;
    __syncthreads();
    float m2 = red[0];
    #pragma unroll
    for (int i = 1; i < 8; ++i) m2 = fmaxf(m2, red[i]);
    __syncthreads();

    float sum = 0.0f;
    #pragma unroll
    for (int i = 0; i < 8; ++i) { float e = __expf(vals[i] - m2); vals[i] = e; sum += e; }
    #pragma unroll
    for (int o = 16; o > 0; o >>= 1) sum += __shfl_xor(sum, o, 32);
    if ((tid & 31) == 0) red[tid >> 5] = sum;
    __syncthreads();
    float tot = 0.0f;
    #pragma unroll
    for (int i = 0; i < 8; ++i) tot += red[i];
    float inv = 1.0f / tot;
    #pragma unroll
    for (int i = 0; i < 8; ++i) {
        int s = tid + i * 256;
        out[b * Sn + s] = vals[i] * inv;
    }
}

// ---------------------------------------------------------------------------
extern "C" void kernel_launch(void* const* d_in, const int* in_sizes, int n_in,
                              void* d_out, int out_size, void* d_ws, size_t ws_size,
                              hipStream_t stream) {
    const float* h      = (const float*)d_in[0];   // (32, 512)
    const float* enc    = (const float*)d_in[1];   // (2048, 32, 1024)
    const int*   mask   = (const int*)  d_in[2];   // (32, 2048)
    const float* attn_w = (const float*)d_in[3];   // (1536, 512)
    const float* attn_b = (const float*)d_in[4];   // (512,)
    const float* v_w    = (const float*)d_in[5];   // (512,)
    float* out = (float*)d_out;                    // (32, 2048)

    char* ws = (char*)d_ws;
    unsigned short* wT     = (unsigned short*)ws;                         // 1 MB
    float*          dec    = (float*)(ws + (1u << 20));                   // 64 KB
    float*          logits = (float*)(ws + (1u << 20) + (64u << 10));     // 256 KB

    // w_enc starts at attn_w row 512 (after w_dec)
    prep_wenc_kernel<<<(Ndim * Kdim) / 256, 256, 0, stream>>>(attn_w + (size_t)Ndim * Ndim, wT);
    prep_dec_kernel<<<(Bn * Ndim) / 256, 256, 0, stream>>>(h, attn_w, attn_b, dec);

    size_t lds_bytes = (size_t)(M_TILE * A_STRIDE + 16 * B_STRIDE) * sizeof(unsigned short);
    bahdanau_energy_kernel<<<Mdim / M_TILE, 256, lds_bytes, stream>>>(enc, wT, dec, v_w, logits);

    softmax_kernel<<<Bn, 256, 0, stream>>>(logits, mask, out);
}
